// QuantizedGPTModel_38491496907083
// MI455X (gfx1250) — matrix operand order repaired and checked by reading, not compiled, not run on hardware
//
#include <hip/hip_runtime.h>
#include <hip/hip_bf16.h>
#include <math.h>
#include <stdint.h>

// Problem constants (match reference)
#define LNUM  8
#define EMB   1024
#define NH    16
#define SEQ   1024
#define VOC   32000
#define BATCH 2
#define HDIM  64
#define MROWS (BATCH*SEQ)   // 2048 token rows

typedef _Float16 h16;
typedef __attribute__((ext_vector_type(16))) _Float16 v16h;
typedef __attribute__((ext_vector_type(8)))  _Float16 v8h;
typedef __attribute__((ext_vector_type(8)))  float    v8f;
typedef __attribute__((ext_vector_type(16))) char     v16c;

// GEMM epilogue modes (compile-time)
#define MODE_F32   0   // outF[m,n] = val                (LM head logits)
#define MODE_F16   1   // outH[m,n] = (f16)val           (qkv)
#define MODE_RESID 2   // resid[m,n] += val              (attn-proj / mlp-proj)
#define MODE_GELU  3   // outH[m,n] = (f16)gelu(val)     (mlp fc)

// -------------------------------------------------------------------------
// Embedding: x[m, c] = tok_emb[idx[m], c] + pos_emb[m % T, c]
// -------------------------------------------------------------------------
__global__ void embed_kernel(const int* __restrict__ idx,
                             const float* __restrict__ tok,
                             const float* __restrict__ pos,
                             float* __restrict__ x)
{
    size_t i = (size_t)blockIdx.x * blockDim.x + threadIdx.x;
    if (i >= (size_t)MROWS * EMB) return;
    int m = (int)(i / EMB);
    int c = (int)(i % EMB);
    int t = m % SEQ;
    int tk = idx[m];
    x[i] = tok[(size_t)tk * EMB + c] + pos[(size_t)t * EMB + c];
}

// -------------------------------------------------------------------------
// LayerNorm (f32 in, f16 out). One 256-thread block per row, E=1024.
// -------------------------------------------------------------------------
__global__ __launch_bounds__(256)
void layernorm_f16(const float* __restrict__ x,
                   const float* __restrict__ g,
                   const float* __restrict__ b,
                   h16* __restrict__ out)
{
    __shared__ float red[256];
    const int m = blockIdx.x;
    const int t = threadIdx.x;
    const float* xr = x + (size_t)m * EMB;

    float v0 = xr[t], v1 = xr[t + 256], v2 = xr[t + 512], v3 = xr[t + 768];
    red[t] = v0 + v1 + v2 + v3;
    __syncthreads();
    for (int st = 128; st > 0; st >>= 1) {
        if (t < st) red[t] += red[t + st];
        __syncthreads();
    }
    const float mean = red[0] * (1.0f / EMB);
    __syncthreads();

    float d0 = v0 - mean, d1 = v1 - mean, d2 = v2 - mean, d3 = v3 - mean;
    red[t] = d0 * d0 + d1 * d1 + d2 * d2 + d3 * d3;
    __syncthreads();
    for (int st = 128; st > 0; st >>= 1) {
        if (t < st) red[t] += red[t + st];
        __syncthreads();
    }
    const float rstd = rsqrtf(red[0] * (1.0f / EMB) + 1e-5f);

    h16* orow = out + (size_t)m * EMB;
    orow[t]       = (h16)(d0 * rstd * g[t]       + b[t]);
    orow[t + 256] = (h16)(d1 * rstd * g[t + 256] + b[t + 256]);
    orow[t + 512] = (h16)(d2 * rstd * g[t + 512] + b[t + 512]);
    orow[t + 768] = (h16)(d3 * rstd * g[t + 768] + b[t + 768]);
}

// -------------------------------------------------------------------------
// V transpose: vT[b,h,d,t] = qkv16[b*T+t, 2E + h*64 + d]  (f16)
// -------------------------------------------------------------------------
__global__ void transpose_v(const h16* __restrict__ qkv, h16* __restrict__ vT)
{
    size_t i = (size_t)blockIdx.x * blockDim.x + threadIdx.x;
    if (i >= (size_t)MROWS * EMB) return;
    int m = (int)(i / EMB);      // b*T + t
    int c = (int)(i % EMB);      // h*64 + d
    int bb = m / SEQ, t = m % SEQ;
    int hh = c / HDIM, d = c % HDIM;
    vT[(((size_t)bb * NH + hh) * HDIM + d) * SEQ + t] =
        qkv[(size_t)m * 3 * EMB + 2 * EMB + c];
}

// -------------------------------------------------------------------------
// WMMA GEMM with double-buffered GLOBAL_LOAD_ASYNC_TO_LDS staging.
// C[M,N] = A[M,K](f16) x W[N,K](int8)^T * scale + bias
//
// Block = 256 threads = 8 wave32s; block tile 64(M) x 128(N); k-step 32.
// Per k-step the block stages:
//   A tile  64x32 f16  = 4 KB  (1x async b128 per thread)
//   W tile 128x32 int8 = 4 KB  (1x async b128 per thread)
// into LDS buffer [buf^1], while WMMAs consume buffer [buf]:
//   s_wait_asynccnt 0 (own loads) -> s_barrier (all loads visible)
//   -> issue next-tile async loads -> ds_load_b128 fragments -> 4 WMMAs.
// A buffer is only rewritten after the *next* barrier, and each wave's
// ds reads are DScnt-complete before it reaches that barrier -> race-free
// with a single barrier per iteration.
// -------------------------------------------------------------------------
template <int MODE, bool HAS_BIAS>
__global__ __launch_bounds__(256)
void gemm_wmma(const h16* __restrict__ A, const int8_t* __restrict__ W,
               const float* __restrict__ scale_p, const float* __restrict__ bias,
               float* __restrict__ outF, h16* __restrict__ outH,
               float* __restrict__ resid,
               int M, int N, int K)
{
    __shared__ __align__(16) h16    ldsA[2][64 * 32];    // 2 x 4 KB
    __shared__ __align__(16) int8_t ldsW[2][128 * 32];   // 2 x 4 KB

    const int tid  = threadIdx.x;
    const int lane = tid & 31;
    const int wave = tid >> 5;
    const int hi   = lane >> 4;
    const int lr   = lane & 15;
    const int wm   = (wave >> 2) * 32;       // wave row base in block tile
    const int wn   = (wave & 3)  * 32;       // wave col base in block tile
    const int mblk = blockIdx.y * 64;
    const int nblk = blockIdx.x * 128;
    const float scale = scale_p[0];

    // cooperative staging assignment (one 16B chunk per thread per tile)
    const int arow = tid >> 2, aseg = tid & 3;   // A: 64 rows x 4 chunks
    const int wrow = tid >> 1, wseg = tid & 1;   // W: 128 rows x 2 chunks
    const h16*    gA = A + (size_t)(mblk + arow) * K + aseg * 8;
    const int8_t* gW = W + (size_t)(nblk + wrow) * K + wseg * 16;

    auto stage = [&](int buf, int kb) {
        uint32_t la = (uint32_t)(uintptr_t)(&ldsA[buf][arow * 32 + aseg * 8]);
        uint64_t ga = (uint64_t)(uintptr_t)(gA + kb);
        asm volatile("global_load_async_to_lds_b128 %0, %1, off"
                     :: "v"(la), "v"(ga) : "memory");
        uint32_t lw = (uint32_t)(uintptr_t)(&ldsW[buf][wrow * 32 + wseg * 16]);
        uint64_t gw = (uint64_t)(uintptr_t)(gW + kb);
        asm volatile("global_load_async_to_lds_b128 %0, %1, off"
                     :: "v"(lw), "v"(gw) : "memory");
    };

    v8f acc[2][2];
    #pragma unroll
    for (int i = 0; i < 2; i++)
        #pragma unroll
        for (int j = 0; j < 2; j++)
            acc[i][j] = (v8f){};

    stage(0, 0);
    int buf = 0;
    for (int kb = 0; kb < K; kb += 32, buf ^= 1) {
        asm volatile("s_wait_asynccnt 0x0" ::: "memory");
        __syncthreads();
        if (kb + 32 < K) stage(buf ^ 1, kb + 32);

        v16h afr[2], bfr[2];
        #pragma unroll
        for (int ti = 0; ti < 2; ti++) {
            const h16* ap = &ldsA[buf][(wm + ti * 16 + lr) * 32];
            union { v16h v; v8h h2[2]; } ua;
            ua.h2[0] = *(const v8h*)(ap + hi * 16);        // K hi*16  .. +7
            ua.h2[1] = *(const v8h*)(ap + hi * 16 + 8);    // K hi*16+8.. +7
            afr[ti] = ua.v;
        }
        #pragma unroll
        for (int tj = 0; tj < 2; tj++) {
            const int8_t* wp = &ldsW[buf][(wn + tj * 16 + lr) * 32 + hi * 16];
            v16c wq = *(const v16c*)wp;                    // ds_load_b128
            v16h bv;
            #pragma unroll
            for (int e = 0; e < 16; e++) bv[e] = (_Float16)(int)(signed char)wq[e];
            bfr[tj] = bv;
        }
        #pragma unroll
        for (int ti = 0; ti < 2; ti++)
            #pragma unroll
            for (int tj = 0; tj < 2; tj++)
                acc[ti][tj] = __builtin_amdgcn_wmma_f32_16x16x32_f16(
                    false, afr[ti], false, bfr[tj],
                    (short)0, acc[ti][tj], false, false);
    }

    // Epilogue. C/D layout: VGPR v holds M = v + 8*hi, N = lane&15.
    const int m0 = mblk + wm;
    const int n0 = nblk + wn;
    #pragma unroll
    for (int ti = 0; ti < 2; ti++) {
        #pragma unroll
        for (int tj = 0; tj < 2; tj++) {
            const int n = n0 + tj * 16 + lr;
            const float bval = HAS_BIAS ? bias[n] : 0.0f;
            #pragma unroll
            for (int v = 0; v < 8; v++) {
                const int m = m0 + ti * 16 + v + 8 * hi;
                const float val = acc[ti][tj][v] * scale + bval;
                const size_t o = (size_t)m * N + n;
                if (MODE == MODE_F32) {
                    outF[o] = val;
                } else if (MODE == MODE_F16) {
                    outH[o] = (h16)val;
                } else if (MODE == MODE_RESID) {
                    resid[o] = resid[o] + val;
                } else { // MODE_GELU: exact 0.5*x*(1+erf(x/sqrt(2)))
                    float gl = 0.5f * val * (1.0f + erff(val * 0.70710678118f));
                    outH[o] = (h16)gl;
                }
            }
        }
    }
}

// -------------------------------------------------------------------------
// WMMA flash attention. One wave32 per 16-query tile of one (b, h).
// S(16x32) = Q x K^T (4 WMMAs) -> causal mask + online softmax
// (shfl_xor row reductions; C-layout row m = v + 8*hi indexes accumulators
// directly) -> P staged via per-wave LDS slab (C->A layout) ->
// O += P x V (4 WMMAs, V^T rows give contiguous per-lane B fragments).
// All control flow is wave-uniform -> EXEC all ones (WMMA requirement).
// -------------------------------------------------------------------------
#define AT_WAVES 8

__global__ __launch_bounds__(256)
void attention_wmma(const h16* __restrict__ qkv, const h16* __restrict__ vT,
                    h16* __restrict__ y)
{
    __shared__ __align__(16) h16 plds[AT_WAVES][16 * 32];
    const int lane = threadIdx.x & 31;
    const int wave = threadIdx.x >> 5;
    const int hi   = lane >> 4;
    const int lr   = lane & 15;

    const int tile = blockIdx.x * AT_WAVES + wave;        // 2048 tiles total
    const int b    = tile / (NH * (SEQ / 16));
    const int rem  = tile % (NH * (SEQ / 16));
    const int h    = rem / (SEQ / 16);
    const int q0   = (rem % (SEQ / 16)) * 16;

    const size_t rs = 3 * EMB;

    v16h qf[2];
    {
        const h16* qrow = qkv + (size_t)(b * SEQ + q0 + lr) * rs + h * HDIM;
        #pragma unroll
        for (int ks = 0; ks < 2; ks++) {
            union { v16h v; v8h h2[2]; } u;
            u.h2[0] = *(const v8h*)(qrow + ks * 32 + hi * 16);
            u.h2[1] = *(const v8h*)(qrow + ks * 32 + hi * 16 + 8);
            qf[ks] = u.v;
        }
    }

    v8f oacc[4];
    #pragma unroll
    for (int d = 0; d < 4; d++) oacc[d] = (v8f){};
    float mrow[8], lrow[8];
    #pragma unroll
    for (int v = 0; v < 8; v++) { mrow[v] = -3.402823e38f; lrow[v] = 0.0f; }

    const h16* kbase = qkv + (size_t)b * SEQ * rs + EMB + h * HDIM;
    const h16* vtb   = vT + (size_t)(b * NH + h) * HDIM * SEQ;   // [HD][T]
    h16* pl = plds[wave];

    const int nblk = (q0 + 16 + 31) / 32;
    for (int blk = 0; blk < nblk; blk++) {
        const int kb = blk * 32;

        v8f sacc[2];
        sacc[0] = (v8f){}; sacc[1] = (v8f){};
        #pragma unroll
        for (int tn = 0; tn < 2; tn++) {
            const h16* krow = kbase + (size_t)(kb + tn * 16 + lr) * rs;
            #pragma unroll
            for (int ks = 0; ks < 2; ks++) {
                union { v16h v; v8h h2[2]; } u;
                u.h2[0] = *(const v8h*)(krow + ks * 32 + hi * 16);
                u.h2[1] = *(const v8h*)(krow + ks * 32 + hi * 16 + 8);
                sacc[tn] = __builtin_amdgcn_wmma_f32_16x16x32_f16(
                    false, qf[ks], false, u.v, (short)0, sacc[tn], false, false);
            }
        }

        float p0[8], p1[8], corr[8];
        #pragma unroll
        for (int v = 0; v < 8; v++) {
            const int q = q0 + v + 8 * hi;
            float s0 = (kb + lr      <= q) ? sacc[0][v] * 0.125f : -3.402823e38f;
            float s1 = (kb + 16 + lr <= q) ? sacc[1][v] * 0.125f : -3.402823e38f;
            float mx = fmaxf(s0, s1);
            #pragma unroll
            for (int off = 1; off < 16; off <<= 1)
                mx = fmaxf(mx, __shfl_xor(mx, off, 32));
            const float mnew = fmaxf(mrow[v], mx);
            corr[v] = __expf(mrow[v] - mnew);
            p0[v]   = __expf(s0 - mnew);
            p1[v]   = __expf(s1 - mnew);
            float ps = p0[v] + p1[v];
            #pragma unroll
            for (int off = 1; off < 16; off <<= 1)
                ps += __shfl_xor(ps, off, 32);
            lrow[v] = lrow[v] * corr[v] + ps;
            mrow[v] = mnew;
        }

        #pragma unroll
        for (int d = 0; d < 4; d++)
            #pragma unroll
            for (int v = 0; v < 8; v++)
                oacc[d][v] *= corr[v];
        #pragma unroll
        for (int v = 0; v < 8; v++) {
            const int m = v + 8 * hi;
            pl[m * 32 + lr]      = (h16)p0[v];
            pl[m * 32 + 16 + lr] = (h16)p1[v];
        }

        v16h pf;
        {
            union { v16h v; v8h h2[2]; } u;
            u.h2[0] = *(const v8h*)(pl + lr * 32 + hi * 16);
            u.h2[1] = *(const v8h*)(pl + lr * 32 + hi * 16 + 8);
            pf = u.v;
        }

        #pragma unroll
        for (int d = 0; d < 4; d++) {
            const h16* vrow = vtb + (size_t)(d * 16 + lr) * SEQ + kb + hi * 16;
            union { v16h v; v8h h2[2]; } u;
            u.h2[0] = *(const v8h*)(vrow);
            u.h2[1] = *(const v8h*)(vrow + 8);
            oacc[d] = __builtin_amdgcn_wmma_f32_16x16x32_f16(
                false, pf, false, u.v, (short)0, oacc[d], false, false);
        }
    }

    #pragma unroll
    for (int v = 0; v < 8; v++) {
        const int m = v + 8 * hi;
        const float inv = 1.0f / lrow[v];
        h16* yp = y + (size_t)(b * SEQ + q0 + m) * EMB + h * HDIM;
        #pragma unroll
        for (int d = 0; d < 4; d++)
            yp[d * 16 + lr] = (h16)(oacc[d][v] * inv);
    }
}

// -------------------------------------------------------------------------
// Host orchestration
// -------------------------------------------------------------------------
extern "C" void kernel_launch(void* const* d_in, const int* in_sizes, int n_in,
                              void* d_out, int out_size, void* d_ws, size_t ws_size,
                              hipStream_t stream)
{
    (void)in_sizes; (void)n_in; (void)out_size; (void)ws_size;

    const int*    idx      = (const int*)   d_in[0];
    const float*  tok_emb  = (const float*) d_in[1];
    const float*  pos_emb  = (const float*) d_in[2];
    const float*  ln1_g    = (const float*) d_in[3];
    const float*  ln1_b    = (const float*) d_in[4];
    const int8_t* attn_qw  = (const int8_t*)d_in[5];
    const float*  attn_s   = (const float*) d_in[6];
    const float*  attn_b   = (const float*) d_in[7];
    const int8_t* aproj_qw = (const int8_t*)d_in[8];
    const float*  aproj_s  = (const float*) d_in[9];
    const float*  aproj_b  = (const float*) d_in[10];
    const float*  ln2_g    = (const float*) d_in[11];
    const float*  ln2_b    = (const float*) d_in[12];
    const int8_t* fc_qw    = (const int8_t*)d_in[13];
    const float*  fc_s     = (const float*) d_in[14];
    const float*  fc_b     = (const float*) d_in[15];
    const int8_t* mproj_qw = (const int8_t*)d_in[16];
    const float*  mproj_s  = (const float*) d_in[17];
    const float*  mproj_b  = (const float*) d_in[18];
    const float*  lnf_g    = (const float*) d_in[19];
    const float*  lnf_b    = (const float*) d_in[20];
    const int8_t* lm_qw    = (const int8_t*)d_in[21];
    const float*  lm_s     = (const float*) d_in[22];

    // Workspace carve-up (48 MiB total)
    uint8_t* ws = (uint8_t*)d_ws;
    float* x     = (float*)(ws);                      //  8 MiB  residual f32 [M,E]
    h16*   hA    = (h16*)  (ws + (size_t)(8  << 20)); //  4 MiB  LN out f16 [M,E]
    h16*   yB    = (h16*)  (ws + (size_t)(12 << 20)); //  4 MiB  attn out f16 [M,E]
    h16*   qkv16 = (h16*)  (ws + (size_t)(16 << 20)); // 12 MiB  qkv f16 [M,3E]
    h16*   vT    = (h16*)  (ws + (size_t)(28 << 20)); //  4 MiB  V^T f16 [B,H,HD,T]
    h16*   g16   = (h16*)  (ws + (size_t)(32 << 20)); // 16 MiB  gelu out f16 [M,4E]

    embed_kernel<<<(MROWS * EMB + 255) / 256, 256, 0, stream>>>(idx, tok_emb, pos_emb, x);

    for (int l = 0; l < LNUM; l++) {
        // ---- attention sublayer ----
        layernorm_f16<<<MROWS, 256, 0, stream>>>(x, ln1_g + (size_t)l * EMB,
                                                 ln1_b + (size_t)l * EMB, hA);
        gemm_wmma<MODE_F16, true><<<dim3(3 * EMB / 128, MROWS / 64), 256, 0, stream>>>(
            hA, attn_qw + (size_t)l * 3 * EMB * EMB, attn_s + l,
            attn_b + (size_t)l * 3 * EMB, nullptr, qkv16, nullptr,
            MROWS, 3 * EMB, EMB);
        transpose_v<<<(MROWS * EMB + 255) / 256, 256, 0, stream>>>(qkv16, vT);
        attention_wmma<<<(BATCH * NH * (SEQ / 16)) / AT_WAVES, 256, 0, stream>>>(
            qkv16, vT, yB);
        gemm_wmma<MODE_RESID, true><<<dim3(EMB / 128, MROWS / 64), 256, 0, stream>>>(
            yB, aproj_qw + (size_t)l * EMB * EMB, aproj_s + l,
            aproj_b + (size_t)l * EMB, nullptr, nullptr, x,
            MROWS, EMB, EMB);

        // ---- MLP sublayer ----
        layernorm_f16<<<MROWS, 256, 0, stream>>>(x, ln2_g + (size_t)l * EMB,
                                                 ln2_b + (size_t)l * EMB, hA);
        gemm_wmma<MODE_GELU, true><<<dim3(4 * EMB / 128, MROWS / 64), 256, 0, stream>>>(
            hA, fc_qw + (size_t)l * 4 * EMB * EMB, fc_s + l,
            fc_b + (size_t)l * 4 * EMB, nullptr, g16, nullptr,
            MROWS, 4 * EMB, EMB);
        gemm_wmma<MODE_RESID, true><<<dim3(EMB / 128, MROWS / 64), 256, 0, stream>>>(
            g16, mproj_qw + (size_t)l * EMB * 4 * EMB, mproj_s + l,
            mproj_b + (size_t)l * EMB, nullptr, nullptr, x,
            MROWS, EMB, 4 * EMB);
    }

    // ---- final LN + LM head ----
    layernorm_f16<<<MROWS, 256, 0, stream>>>(x, lnf_g, lnf_b, hA);
    gemm_wmma<MODE_F32, false><<<dim3(VOC / 128, MROWS / 64), 256, 0, stream>>>(
        hA, lm_qw, lm_s, nullptr, (float*)d_out, nullptr, nullptr,
        MROWS, VOC, EMB);
}
